// Tpseg_81441169867241
// MI455X (gfx1250) — matrix operand order repaired and checked by reading, not compiled, not run on hardware
//
#include <hip/hip_runtime.h>
#include <hip/hip_bf16.h>

// ---------------------------------------------------------------------------
// MI455X (gfx1250) proto-segmentation forward pass.
// 1x1-conv / K-V projection stages run on v_wmma_f32_16x16x32_f16:
//   - block = Cout/16 waves; CINx16 activation tile converted to packed f16
//     and staged once in LDS (coalesced global loads, conflict-free layout)
//   - wave w owns out-channel tile w; weight fragments hoisted in registers
//   - bias hoisted per-lane before the tile loop (branch-free epilogue)
//   - D stores coalesced along the spatial lane dimension
// All shape divisions are compile-time (HW = 16384).
// ---------------------------------------------------------------------------

typedef __attribute__((ext_vector_type(16))) _Float16     v16h;
typedef __attribute__((ext_vector_type(2)))  _Float16     v2h;
typedef __attribute__((ext_vector_type(8)))  float        v8f;
typedef __attribute__((ext_vector_type(8)))  unsigned int v8u;

constexpr int BN  = 4;            // batch
constexpr int CH  = 128;          // main channel count
constexpr int HH  = 128;
constexpr int WWD = 128;
constexpr int HWC = HH * WWD;     // 16384 (power of two: >>14, &16383)
constexpr long FSZ = (long)BN * CH * HWC;   // full activation, floats
constexpr int SP_TILES = (BN * HWC) / 16;   // 4096 spatial 16-row tiles
constexpr int NT_PER_BLK = 8;               // spatial tiles per block
constexpr int C1X1_GRID = SP_TILES / NT_PER_BLK;  // 512 blocks

__device__ __forceinline__ float gelu_exact(float x) {
    return 0.5f * x * (1.0f + erff(x * 0.70710678118654752440f));
}
__device__ __forceinline__ float sigmoid_f(float x) {
    return 1.0f / (1.0f + __expf(-x));
}

// A-matrix 16x32 f16 layout (ISA 7.12.2): lanes 0-15/16-31 both cover M=idx;
// e<8 -> K = e + 8*half ; e>=8 -> K = (e-8) + 16 + 8*half
__device__ __forceinline__ int kmapA(int e, int half) {
    return (e < 8) ? (e + 8 * half) : (e + 8 + 8 * half);
}

// ---------------------------------------------------------------------------
// WMMA 1x1 conv (NCHW): out[b,o,p] = sum_k act[b,k,p] * W[o,k] + bias[o]
// LDS holds packed f16: dword [kp*18 + sp] = halves (k=2kp, k=2kp+1), col sp.
// Stride 18 dwords -> lanes 0-15 and 16-31 hit disjoint LDS bank sets.
// ---------------------------------------------------------------------------
template<int CIN, int C0, int COUT>
__global__ void __launch_bounds__((COUT / 16) * 32)
conv1x1_wmma_kernel(const float* __restrict__ in0, const float* __restrict__ in1,
                    const float* __restrict__ W, const float* __restrict__ bias,
                    float* __restrict__ out)
{
    constexpr int KF   = CIN / 32;       // K fragments per tile
    constexpr int NW   = COUT / 16;      // waves per block = channel tiles
    constexpr int NTHR = NW * 32;
    constexpr int PSTR = 18;             // LDS pair-row stride (dwords)
    constexpr int EPT  = ((CIN / 2) * 16) / NTHR;  // loader dwords per thread
    constexpr int C1   = CIN - C0;

    __shared__ unsigned int ldsu[(CIN / 2) * PSTR];

    const int tid  = threadIdx.x;
    const int lane = tid & 31;
    const int wv   = tid >> 5;           // channel tile owned by this wave
    const int half = lane >> 4;
    const int idx  = lane & 15;
    const int ch0  = wv * 16;

    // Hoisted weight fragments: A rows = out-channels ch0+idx (per-lane row).
    const float* wr = W + (long)(ch0 + idx) * CIN;
    v16h wfrag[KF];
#pragma unroll
    for (int f = 0; f < KF; ++f) {
#pragma unroll
        for (int e = 0; e < 16; ++e)
            wfrag[f][e] = (_Float16)wr[f * 32 + kmapA(e, half)];
    }

    // Hoisted bias (loop-invariant): row r of this lane is channel ch0+r+8*half.
    float bv[8];
#pragma unroll
    for (int r = 0; r < 8; ++r)
        bv[r] = bias ? bias[ch0 + r + 8 * half] : 0.0f;

    const int t0 = blockIdx.x * NT_PER_BLK;
#pragma unroll 1
    for (int s = 0; s < NT_PER_BLK; ++s) {
        const long m0 = (long)(t0 + s) * 16;
        const int  b  = (int)(m0 >> 14);       // HWC == 16384
        const int  p0 = (int)(m0 & 16383);
        const float* src0 = in0 + (long)b * C0 * HWC + p0;
        const float* src1 = (C1 > 0) ? (in1 + (long)b * C1 * HWC + p0) : src0;

        // Cooperative fill: each thread handles a (k-pair, sp) cell:
        // two sp-coalesced global loads, one cvt_pk, one ds_store_b32.
        // (C0 is even, so a k-pair never straddles the concat boundary.)
#pragma unroll
        for (int j = 0; j < EPT; ++j) {
            const int flat = j * NTHR + tid;
            const int kp = flat >> 4;
            const int sp = flat & 15;
            const int k0 = kp * 2;
            float va, vb;
            if (k0 < C0) {
                va = src0[(long)k0 * HWC + sp];
                vb = src0[(long)(k0 + 1) * HWC + sp];
            } else {
                va = src1[(long)(k0 - C0) * HWC + sp];
                vb = src1[(long)(k0 - C0 + 1) * HWC + sp];
            }
            v2h hv = { (_Float16)va, (_Float16)vb };
            ldsu[kp * PSTR + sp] = __builtin_bit_cast(unsigned int, hv);
        }
        __syncthreads();

        v8f acc = {};
#pragma unroll
        for (int f = 0; f < KF; ++f) {
            // v16h dword d holds k = f*32 + 2d + 8*half (+8 if d>=4) (+1) at column idx.
            v8u bw;
#pragma unroll
            for (int d = 0; d < 8; ++d)
                bw[d] = ldsu[(f * 16 + d + 4 * half + (d >= 4 ? 4 : 0)) * PSTR + idx];
            v16h bfrag = __builtin_bit_cast(v16h, bw);
            acc = __builtin_amdgcn_wmma_f32_16x16x32_f16(
                /*neg_a=*/false, wfrag[f], /*neg_b=*/false, bfrag,
                /*c_mod=*/(short)0, acc, /*reuse_a=*/false, /*reuse_b=*/false);
        }

        // D: row = vgpr + 8*half (out-channel), col = idx (spatial, coalesced)
        float* ob = out + ((long)b * COUT + ch0 + 8 * half) * HWC + p0 + idx;
#pragma unroll
        for (int r = 0; r < 8; ++r)
            ob[r * HWC] = acc[r] + bv[r];   // constant offsets -> store IOFFSET
        __syncthreads();   // LDS reused by next tile
    }
}

// ---------------------------------------------------------------------------
// Bilinear resize matching jnp.linspace(0, h-1, oh) sampling (align-corners).
// ---------------------------------------------------------------------------
__global__ void resize_bilinear_kernel(const float* __restrict__ in, float* __restrict__ out,
                                       int BC, int h, int w, int oh, int ow)
{
    long i = (long)blockIdx.x * blockDim.x + threadIdx.x;
    long total = (long)BC * oh * ow;
    if (i >= total) return;
    int ox = (int)(i % ow); long t = i / ow;
    int oy = (int)(t % oh); int bc = (int)(t / oh);
    float fy = (oh > 1) ? (float)oy * (float)(h - 1) / (float)(oh - 1) : 0.0f;
    float fx = (ow > 1) ? (float)ox * (float)(w - 1) / (float)(ow - 1) : 0.0f;
    int y0 = (int)floorf(fy); int y1 = min(y0 + 1, h - 1);
    int x0 = (int)floorf(fx); int x1 = min(x0 + 1, w - 1);
    float wy = fy - (float)y0, wx = fx - (float)x0;
    const float* s = in + (long)bc * h * w;
    float tl = s[y0 * w + x0], tr = s[y0 * w + x1];
    float bl = s[y1 * w + x0], br = s[y1 * w + x1];
    out[i] = tl * (1 - wy) * (1 - wx) + tr * (1 - wy) * wx
           + bl * wy * (1 - wx)       + br * wy * wx;
}

// ---------------------------------------------------------------------------
// GroupNorm: per-(batch, group) mean / inv-std, then affine (+ optional GELU).
// ---------------------------------------------------------------------------
__global__ void __launch_bounds__(256)
gn_stats_kernel(const float* __restrict__ x, float* __restrict__ stats, int cpg, int G)
{
    int bg = blockIdx.x; int b = bg / G, g = bg % G;
    long n = (long)cpg * HWC;
    const float* p = x + ((long)b * (cpg * G) + (long)g * cpg) * HWC;
    __shared__ float s1[256], s2[256];
    float a = 0.f, q = 0.f;
    for (long i = threadIdx.x; i < n; i += 256) { float v = p[i]; a += v; q += v * v; }
    s1[threadIdx.x] = a; s2[threadIdx.x] = q; __syncthreads();
    for (int st = 128; st > 0; st >>= 1) {
        if (threadIdx.x < st) { s1[threadIdx.x] += s1[threadIdx.x + st]; s2[threadIdx.x] += s2[threadIdx.x + st]; }
        __syncthreads();
    }
    if (threadIdx.x == 0) {
        float mu = s1[0] / (float)n;
        float var = s2[0] / (float)n - mu * mu;
        stats[bg * 2] = mu;
        stats[bg * 2 + 1] = rsqrtf(fmaxf(var, 0.0f) + 1e-5f);
    }
}

__global__ void gn_apply_kernel(float* __restrict__ x, const float* __restrict__ gamma,
                                const float* __restrict__ beta, const float* __restrict__ stats,
                                int C, int cpg, int G, int act)
{
    long i = (long)blockIdx.x * blockDim.x + threadIdx.x;
    long total = (long)BN * C * HWC;
    if (i >= total) return;
    int c = (int)((i >> 14) % C);
    int b = (int)(i / ((long)C * HWC));
    int g = c / cpg;
    float mu = stats[(b * G + g) * 2], is = stats[(b * G + g) * 2 + 1];
    float v = (x[i] - mu) * is * gamma[c] + beta[c];
    x[i] = act ? gelu_exact(v) : v;
}

// ---------------------------------------------------------------------------
// Grouped 3x3 conv (pad=1), Cin=Cout=128. enc: groups=32; head: depthwise 128.
// ---------------------------------------------------------------------------
__global__ void conv3x3_grouped_kernel(const float* __restrict__ in, const float* __restrict__ w,
                                       const float* __restrict__ bias, float* __restrict__ out,
                                       int groups)
{
    long i = (long)blockIdx.x * blockDim.x + threadIdx.x;
    if (i >= FSZ) return;
    int xq = (int)(i & (WWD - 1));
    int yq = (int)((i >> 7) & (HH - 1));
    int o  = (int)((i >> 14) & (CH - 1));
    int b  = (int)(i >> 21);
    int ipg = CH / groups, go = o / ipg;   // opg == ipg for both uses (4,4) (1,1)
    float acc = bias ? bias[o] : 0.0f;
    for (int ig = 0; ig < ipg; ++ig) {
        const float* xi = in + ((long)b * CH + go * ipg + ig) * HWC;
        const float* wk = w + ((long)o * ipg + ig) * 9;
        for (int ky = 0; ky < 3; ++ky) {
            int yy = yq + ky - 1; if (yy < 0 || yy >= HH) continue;
            for (int kx = 0; kx < 3; ++kx) {
                int xx = xq + kx - 1; if (xx < 0 || xx >= WWD) continue;
                acc += wk[ky * 3 + kx] * xi[yy * WWD + xx];
            }
        }
    }
    out[i] = acc;
}

// ---------------------------------------------------------------------------
// Prototype init MLP (batch 0 only; broadcast): fg/bg = l2norm halves of pi.
// ---------------------------------------------------------------------------
__global__ void __launch_bounds__(256)
pinit_kernel(const float* __restrict__ te, const float* __restrict__ w1, const float* __restrict__ b1,
             const float* __restrict__ w2, const float* __restrict__ b2, float* __restrict__ fgbg)
{
    __shared__ float hid[256];
    __shared__ float pi[256];
    __shared__ float nrm[2];
    int tid = threadIdx.x;
    float s = b1[tid];
    for (int t = 0; t < 64; ++t) s += te[t] * w1[tid * 64 + t];
    hid[tid] = gelu_exact(s);
    __syncthreads();
    float o = b2[tid];
    for (int t = 0; t < 256; ++t) o += hid[t] * w2[tid * 256 + t];
    pi[tid] = o;
    __syncthreads();
    if (tid < 2) {
        float ss = 0.f;
        for (int c = 0; c < 128; ++c) { float v = pi[tid * 128 + c]; ss += v * v; }
        nrm[tid] = fmaxf(sqrtf(ss), 1e-12f);
    }
    __syncthreads();
    fgbg[tid] = pi[tid] / nrm[tid >> 7];
}

// fused_sim[b,p] = <l2norm_c(sf), fg - bg>
__global__ void simfused_kernel(const float* __restrict__ sf, const float* __restrict__ fgbg,
                                float* __restrict__ fsim)
{
    long i = (long)blockIdx.x * blockDim.x + threadIdx.x;
    if (i >= (long)BN * HWC) return;
    int p = (int)(i & (HWC - 1)), b = (int)(i >> 14);
    const float* s = sf + (long)b * CH * HWC + p;
    float nrm = 0.f;
    for (int c = 0; c < CH; ++c) { float v = s[(long)c * HWC]; nrm += v * v; }
    float inv = 1.0f / fmaxf(sqrtf(nrm), 1e-12f);
    float acc = 0.f;
    for (int c = 0; c < CH; ++c) acc += s[(long)c * HWC] * inv * (fgbg[c] - fgbg[128 + c]);
    fsim[i] = acc;
}

// Generic small linear: out[r,n] = act(in[r,:] . w[n,:] + bias[n])
__global__ void linear_kernel(const float* __restrict__ in, const float* __restrict__ w,
                              const float* __restrict__ bias, float* __restrict__ out,
                              int rows, int K, int N, int act)
{
    long i = (long)blockIdx.x * blockDim.x + threadIdx.x;
    if (i >= (long)rows * N) return;
    int r = (int)(i / N), n = (int)(i % N);
    float acc = bias ? bias[n] : 0.0f;
    const float* a = in + (long)r * K;
    const float* ww = w + (long)n * K;
    for (int k = 0; k < K; ++k) acc += a[k] * ww[k];
    out[i] = act ? gelu_exact(acc) : acc;
}

// ---------------------------------------------------------------------------
// Attention: one block per (b, head, query). 2 queries x 16384 keys x dh=32.
// ---------------------------------------------------------------------------
__global__ void __launch_bounds__(256)
attn_kernel(const float* __restrict__ Q, const float* __restrict__ K, const float* __restrict__ V,
            float* __restrict__ scores, float* __restrict__ attnout)
{
    int blk = blockIdx.x;                      // b*8 + h*2 + q
    int b = blk >> 3, h = (blk >> 1) & 3, q = blk & 1;
    int tid = threadIdx.x;
    const float scale = 0.17677669529663687f;  // 1/sqrt(32)
    const float* Kb = K + (long)b * CH * HWC + (long)h * 32 * HWC;
    const float* Vb = V + (long)b * CH * HWC + (long)h * 32 * HWC;
    const float* Qv = Q + q * 128 + h * 32;
    float* sc = scores + (long)blk * HWC;
    __shared__ float red[256];
    float qr[32];
#pragma unroll
    for (int d = 0; d < 32; ++d) qr[d] = Qv[d];
    float lmax = -1e30f;
    for (int t = tid; t < HWC; t += 256) {
        float s = 0.f;
        for (int d = 0; d < 32; ++d) s += qr[d] * Kb[(long)d * HWC + t];
        s *= scale;
        sc[t] = s;
        lmax = fmaxf(lmax, s);
    }
    red[tid] = lmax; __syncthreads();
    for (int st = 128; st > 0; st >>= 1) { if (tid < st) red[tid] = fmaxf(red[tid], red[tid + st]); __syncthreads(); }
    float gmax = red[0]; __syncthreads();
    float lsum = 0.f;
    for (int t = tid; t < HWC; t += 256) { float e = __expf(sc[t] - gmax); sc[t] = e; lsum += e; }
    red[tid] = lsum; __syncthreads();
    for (int st = 128; st > 0; st >>= 1) { if (tid < st) red[tid] += red[tid + st]; __syncthreads(); }
    float inv = 1.0f / red[0]; __syncthreads();
    float acc[32];
#pragma unroll
    for (int d = 0; d < 32; ++d) acc[d] = 0.f;
    for (int t = tid; t < HWC; t += 256) {
        float pv = sc[t] * inv;
        for (int d = 0; d < 32; ++d) acc[d] += pv * Vb[(long)d * HWC + t];
    }
    for (int d = 0; d < 32; ++d) {
        red[tid] = acc[d]; __syncthreads();
        for (int st = 128; st > 0; st >>= 1) { if (tid < st) red[tid] += red[tid + st]; __syncthreads(); }
        if (tid == 0) attnout[((long)b * 2 + q) * 128 + h * 32 + d] = red[0];
        __syncthreads();
    }
}

// proto_feat[b] = concat(fgA, bgA, fgA-bgA) @ fuser_w.T + fuser_b
__global__ void protofeat_kernel(const float* __restrict__ att, const float* __restrict__ w,
                                 const float* __restrict__ bias, float* __restrict__ pf)
{
    int i = blockIdx.x * blockDim.x + threadIdx.x;
    if (i >= 4 * 128) return;
    int b = i / 128, o = i % 128;
    const float* fa = att + (long)b * 256;
    const float* ba = fa + 128;
    const float* ww = w + (long)o * 384;
    float acc = bias[o];
    for (int c = 0; c < 128; ++c)
        acc += fa[c] * ww[c] + ba[c] * ww[128 + c] + (fa[c] - ba[c]) * ww[256 + c];
    pf[i] = acc;
}

// expert weights: ew[b,:] = softmax(proto_feat[b] @ wg_w.T + wg_b)
__global__ void ew_kernel(const float* __restrict__ pf, const float* __restrict__ wgw,
                          const float* __restrict__ wgb, float* __restrict__ ew)
{
    __shared__ float l[24];
    int tid = threadIdx.x;
    if (tid < 24) {
        int b = tid / 6, m = tid % 6;
        float s = wgb[m];
        for (int c = 0; c < 128; ++c) s += pf[b * 128 + c] * wgw[m * 128 + c];
        l[tid] = s;
    }
    __syncthreads();
    if (tid < 4) {
        float mx = -1e30f;
        for (int m = 0; m < 6; ++m) mx = fmaxf(mx, l[tid * 6 + m]);
        float e[6]; float sm = 0.f;
        for (int m = 0; m < 6; ++m) { e[m] = __expf(l[tid * 6 + m] - mx); sm += e[m]; }
        for (int m = 0; m < 6; ++m) ew[tid * 6 + m] = e[m] / sm;
    }
}

// k_eff[b,c,r] = s * sum_m ew[b,m]*ek[b,m,c,r] + (1-s)*[r==4]  (sum_m ew == 1)
__global__ void keff_kernel(const float* __restrict__ ew, const float* __restrict__ ekraw,
                            const float* __restrict__ sptr, float* __restrict__ keff)
{
    int i = blockIdx.x * blockDim.x + threadIdx.x;
    if (i >= 4 * 128 * 9) return;
    int r = i % 9, c = (i / 9) % 128, b = i / 1152;
    float s = sptr[0];
    float acc = 0.f;
    for (int m = 0; m < 6; ++m) acc += ew[b * 6 + m] * ekraw[(long)b * 6912 + m * 1152 + c * 9 + r];
    keff[i] = acc * s + (r == 4 ? (1.0f - s) : 0.0f);
}

// y[b,c,h,w] = sum_r keff[b,c,r] * gate(src) * x(src), gate = 1 + 0.5*sigmoid(fsim)
__global__ void dynconv_kernel(const float* __restrict__ x, const float* __restrict__ fsim,
                               const float* __restrict__ keff, float* __restrict__ y)
{
    long i = (long)blockIdx.x * blockDim.x + threadIdx.x;
    if (i >= FSZ) return;
    int wq = (int)(i & (WWD - 1));
    int hq = (int)((i >> 7) & (HH - 1));
    int c  = (int)((i >> 14) & (CH - 1));
    int b  = (int)(i >> 21);
    const float* kb = keff + ((long)b * CH + c) * 9;
    const float* xb = x + ((long)b * CH + c) * HWC;
    const float* sb = fsim + (long)b * HWC;
    float acc = 0.f;
    for (int dy = -1; dy <= 1; ++dy) {
        int hh = hq + dy; if (hh < 0 || hh >= HH) continue;
        for (int dx = -1; dx <= 1; ++dx) {
            int ww = wq + dx; if (ww < 0 || ww >= WWD) continue;
            float g = 1.0f + 0.5f * sigmoid_f(sb[hh * WWD + ww]);
            acc += kb[(dy + 1) * 3 + (dx + 1)] * xb[hh * WWD + ww] * g;
        }
    }
    y[i] = acc;
}

// head 1x1 pw conv to 1 channel
__global__ void headpw_kernel(const float* __restrict__ z, const float* __restrict__ w,
                              const float* __restrict__ bias, float* __restrict__ logits)
{
    long i = (long)blockIdx.x * blockDim.x + threadIdx.x;
    if (i >= (long)BN * HWC) return;
    int b = (int)(i >> 14), p = (int)(i & (HWC - 1));
    float acc = bias[0];
    for (int c = 0; c < CH; ++c) acc += z[((long)b * CH + c) * HWC + p] * w[c];
    logits[i] = acc;
}

__global__ void __launch_bounds__(256)
absmean_kernel(const float* __restrict__ fsim, float* __restrict__ outm, int n)
{
    __shared__ float red[256];
    float s = 0.f;
    for (int i = threadIdx.x; i < n; i += 256) s += fabsf(fsim[i]);
    red[threadIdx.x] = s; __syncthreads();
    for (int st = 128; st > 0; st >>= 1) { if (threadIdx.x < st) red[threadIdx.x] += red[threadIdx.x + st]; __syncthreads(); }
    if (threadIdx.x == 0) outm[0] = red[0] / (float)n;
}

__global__ void logit_adjust_kernel(float* __restrict__ logits, const float* __restrict__ fsim,
                                    const float* __restrict__ meanp, const float* __restrict__ sim_temp,
                                    const float* __restrict__ proto_amp, int n)
{
    int i = blockIdx.x * blockDim.x + threadIdx.x;
    if (i >= n) return;
    float temp = fmaxf(sim_temp[0], 0.001f);
    float sn = fsim[i] / (meanp[0] + 1e-6f);
    logits[i] += 5.0f * proto_amp[0] * tanhf(sn / temp);
}

// fw = tw*scale normalized over a length-1 axis == exactly 1.0
__global__ void fw_kernel(float* __restrict__ fw)
{
    if (threadIdx.x < 4) fw[threadIdx.x] = 1.0f;
}

// ---------------------------------------------------------------------------
extern "C" void kernel_launch(void* const* d_in, const int* in_sizes, int n_in,
                              void* d_out, int out_size, void* d_ws, size_t ws_size,
                              hipStream_t stream)
{
    // ---- inputs -----------------------------------------------------------
    const float* x_low    = (const float*)d_in[0];
    const float* x_high   = (const float*)d_in[1];
    const float* task_emb = (const float*)d_in[2];
    // d_in[3] task_ids: all zeros and logits have 1 channel -> unused
    const float* proj_low_w  = (const float*)d_in[4];
    const float* proj_low_g  = (const float*)d_in[5];
    const float* proj_low_b  = (const float*)d_in[6];
    const float* proj_high_w = (const float*)d_in[7];
    const float* proj_high_g = (const float*)d_in[8];
    const float* proj_high_b = (const float*)d_in[9];
    const float* fuse_w      = (const float*)d_in[10];
    const float* fuse_g      = (const float*)d_in[11];
    const float* fuse_b      = (const float*)d_in[12];
    const float* pinit_w1    = (const float*)d_in[13];
    const float* pinit_b1    = (const float*)d_in[14];
    const float* pinit_w2    = (const float*)d_in[15];
    const float* pinit_b2    = (const float*)d_in[16];
    const float* enc_dw_w    = (const float*)d_in[17];
    const float* enc_dw_b    = (const float*)d_in[18];
    const float* enc_gn_g    = (const float*)d_in[19];
    const float* enc_gn_b    = (const float*)d_in[20];
    const float* enc_pw_w    = (const float*)d_in[21];
    const float* enc_pw_b    = (const float*)d_in[22];
    const float* attn_in_w   = (const float*)d_in[23];
    const float* attn_in_b   = (const float*)d_in[24];
    const float* attn_out_w  = (const float*)d_in[25];
    const float* attn_out_b  = (const float*)d_in[26];
    const float* fuser_w     = (const float*)d_in[27];
    const float* fuser_b     = (const float*)d_in[28];
    // 29..33: router / adaptive_scale -> fw is exactly 1.0, skip
    const float* eg_w1       = (const float*)d_in[34];
    const float* eg_b1       = (const float*)d_in[35];
    const float* eg_w2       = (const float*)d_in[36];
    const float* eg_b2       = (const float*)d_in[37];
    const float* wg_w        = (const float*)d_in[38];
    const float* wg_b        = (const float*)d_in[39];
    const float* exp_scale   = (const float*)d_in[40];
    const float* sim_temp    = (const float*)d_in[41];
    const float* proto_amp   = (const float*)d_in[42];
    const float* head_dw_w   = (const float*)d_in[43];
    const float* head_dw_b   = (const float*)d_in[44];
    const float* head_gn_g   = (const float*)d_in[45];
    const float* head_gn_b   = (const float*)d_in[46];
    const float* head_pw_w   = (const float*)d_in[47];
    const float* head_pw_b   = (const float*)d_in[48];

    // ---- outputs: [pred (4*1*512*512) | y (4*128*128*128) | fw (4)] -------
    float* out_pred = (float*)d_out;
    float* out_y    = out_pred + (long)4 * 512 * 512;
    float* out_fw   = out_y + FSZ;

    // ---- workspace layout (floats) ---------------------------------------
    float* ws = (float*)d_ws;
    float* bufA = ws;
    float* bufB = ws + FSZ;
    float* bufC = ws + 2 * FSZ;
    float* bufD = ws + 3 * FSZ;
    float* fsim   = ws + 4 * FSZ;          // 65536
    float* logits = fsim + 65536;          // 65536
    float* scores = logits + 65536;        // 524288 (32 blocks x 16384)
    float* smallb = scores + 524288;
    float* Qbuf   = smallb;          // 256
    float* fgbg   = smallb + 256;    // 256
    float* attno  = smallb + 512;    // 1024
    float* attnd  = smallb + 1536;   // 1024
    float* pf     = smallb + 2560;   // 512
    float* h1     = smallb + 3072;   // 1024
    float* ewb    = smallb + 4096;   // 24
    float* keff   = smallb + 4224;   // 4608
    float* gnst   = smallb + 8832;   // up to 32
    float* smean  = smallb + 8896;   // 1
    float* ekraw  = smallb + 16384;  // 27648

    auto NB = [](long n) { return (int)((n + 255) / 256); };

    // 1) upsample x_high 64x64 -> 128x128 (bufA)
    resize_bilinear_kernel<<<NB((long)BN * 128 * 128 * 128), 256, 0, stream>>>(
        x_high, bufA, BN * 128, 64, 64, 128, 128);

    // 2) proj_high 1x1 (WMMA) -> bufB ; GN(1)+GELU
    conv1x1_wmma_kernel<128, 128, 128><<<C1X1_GRID, 256, 0, stream>>>(
        bufA, nullptr, proj_high_w, nullptr, bufB);
    gn_stats_kernel<<<BN, 256, 0, stream>>>(bufB, gnst, 128, 1);
    gn_apply_kernel<<<NB(FSZ), 256, 0, stream>>>(bufB, proj_high_g, proj_high_b, gnst, 128, 128, 1, 1);

    // 3) proj_low 1x1 (WMMA) -> bufC ; GN(1)+GELU
    conv1x1_wmma_kernel<64, 64, 64><<<C1X1_GRID, 128, 0, stream>>>(
        x_low, nullptr, proj_low_w, nullptr, bufC);
    gn_stats_kernel<<<BN, 256, 0, stream>>>(bufC, gnst, 64, 1);
    gn_apply_kernel<<<NB(FSZ / 2), 256, 0, stream>>>(bufC, proj_low_g, proj_low_b, gnst, 64, 64, 1, 1);

    // 4) fuse 1x1 over concat[xh(128), xl(64)] (WMMA) -> bufD ; GN(1)+GELU
    conv1x1_wmma_kernel<192, 128, 128><<<C1X1_GRID, 256, 0, stream>>>(
        bufB, bufC, fuse_w, nullptr, bufD);
    gn_stats_kernel<<<BN, 256, 0, stream>>>(bufD, gnst, 128, 1);
    gn_apply_kernel<<<NB(FSZ), 256, 0, stream>>>(bufD, fuse_g, fuse_b, gnst, 128, 128, 1, 1);

    // 5) encoder: grouped dw 3x3 -> bufA ; GN(4)+GELU ; pw 1x1 (WMMA) -> bufC
    conv3x3_grouped_kernel<<<NB(FSZ), 256, 0, stream>>>(bufD, enc_dw_w, enc_dw_b, bufA, 32);
    gn_stats_kernel<<<BN * 4, 256, 0, stream>>>(bufA, gnst, 32, 4);
    gn_apply_kernel<<<NB(FSZ), 256, 0, stream>>>(bufA, enc_gn_g, enc_gn_b, gnst, 128, 32, 4, 1);
    conv1x1_wmma_kernel<128, 128, 128><<<C1X1_GRID, 256, 0, stream>>>(
        bufA, nullptr, enc_pw_w, enc_pw_b, bufC);

    // 6) prototypes (batch 0, broadcast) + fused similarity map
    pinit_kernel<<<1, 256, 0, stream>>>(task_emb, pinit_w1, pinit_b1, pinit_w2, pinit_b2, fgbg);
    simfused_kernel<<<NB((long)BN * HWC), 256, 0, stream>>>(bufC, fgbg, fsim);

    // 7) attention: Q (tiny), K/V projections (WMMA), softmax-attend, out-proj
    linear_kernel<<<1, 256, 0, stream>>>(fgbg, attn_in_w, attn_in_b, Qbuf, 2, 128, 128, 0);
    conv1x1_wmma_kernel<128, 128, 128><<<C1X1_GRID, 256, 0, stream>>>(
        bufC, nullptr, attn_in_w + 128 * 128, attn_in_b + 128, bufA);
    conv1x1_wmma_kernel<128, 128, 128><<<C1X1_GRID, 256, 0, stream>>>(
        bufC, nullptr, attn_in_w + 2 * 128 * 128, attn_in_b + 256, bufB);
    attn_kernel<<<32, 256, 0, stream>>>(Qbuf, bufA, bufB, scores, attno);
    linear_kernel<<<4, 256, 0, stream>>>(attno, attn_out_w, attn_out_b, attnd, 8, 128, 128, 0);
    protofeat_kernel<<<2, 256, 0, stream>>>(attnd, fuser_w, fuser_b, pf);

    // 8) expert kernel generation
    linear_kernel<<<4, 256, 0, stream>>>(pf, eg_w1, eg_b1, h1, 4, 128, 256, 1);
    linear_kernel<<<NB(4L * 6912), 256, 0, stream>>>(h1, eg_w2, eg_b2, ekraw, 4, 256, 6912, 0);
    ew_kernel<<<1, 32, 0, stream>>>(pf, wg_w, wg_b, ewb);
    keff_kernel<<<NB(4L * 128 * 9), 256, 0, stream>>>(ewb, ekraw, exp_scale, keff);

    // 9) gated dynamic 9-tap conv -> y (second output)
    dynconv_kernel<<<NB(FSZ), 256, 0, stream>>>(bufD, fsim, keff, out_y);

    // 10) head: dw 3x3 -> bufA ; GN(1)+GELU ; pw to 1 channel -> logits
    conv3x3_grouped_kernel<<<NB(FSZ), 256, 0, stream>>>(out_y, head_dw_w, head_dw_b, bufA, 128);
    gn_stats_kernel<<<BN, 256, 0, stream>>>(bufA, gnst, 128, 1);
    gn_apply_kernel<<<NB(FSZ), 256, 0, stream>>>(bufA, head_gn_g, head_gn_b, gnst, 128, 128, 1, 1);
    headpw_kernel<<<NB((long)BN * HWC), 256, 0, stream>>>(bufA, head_pw_w, head_pw_b, logits);

    // 11) prototype-similarity residual + 4x upsample -> pred (first output)
    absmean_kernel<<<1, 256, 0, stream>>>(fsim, smean, BN * HWC);
    logit_adjust_kernel<<<NB((long)BN * HWC), 256, 0, stream>>>(
        logits, fsim, smean, sim_temp, proto_amp, BN * HWC);
    resize_bilinear_kernel<<<NB((long)BN * 512 * 512), 256, 0, stream>>>(
        logits, out_pred, BN, 128, 128, 512, 512);

    // 12) fw == 1.0 exactly (softmax over singleton axis, self-normalized)
    fw_kernel<<<1, 32, 0, stream>>>(out_fw);
}
